// PointNet2_80917183857078
// MI455X (gfx1250) — compile-verified, matrix operand order checked by reading, not run on hardware
//
#include <hip/hip_runtime.h>

typedef __attribute__((ext_vector_type(16))) _Float16 v16h;
typedef __attribute__((ext_vector_type(8)))  _Float16 v8h;
typedef __attribute__((ext_vector_type(8)))  float    v8f;

#define DEVINL __device__ __forceinline__

// ---------------------------------------------------------------------------
// WMMA helpers (CDNA5 wave32, V_WMMA_F32_16X16X32_F16)
// A: 16x32 f16 (M on lanes 0..15 / 16..31, K packed per doc layout)
// B: 32x16 f16 (N on lanes, mirrored K layout) -- weights pre-swizzled so each
//    lane reads 16 contiguous halves (32B).
// C/D: 16x16 f32, VGPR j -> row (j + 8*(lane/16)), col (lane%16)
// ---------------------------------------------------------------------------
DEVINL v8f wmma_f32_f16(v16h a, v16h b, v8f c) {
  return __builtin_amdgcn_wmma_f32_16x16x32_f16(
      /*neg_a=*/false, a, /*neg_b=*/false, b,
      /*c_mod=*/(short)0, c, /*reuse_a=*/false, /*reuse_b=*/false);
}

// A-fragment: rows mtile*16 + lane%16 of src[M][ldk] (f16, row-major), k-block ks.
// lane half h -> k = ks*32 + (h&8)*2 + (lane>>4)*8 + (h&7)  => two 16B loads.
DEVINL v16h load_afrag(const _Float16* src, int ldk, int mtile, int ks, int lane) {
  int m = mtile * 16 + (lane & 15);
  int g = lane >> 4;
  const _Float16* p = src + (size_t)m * ldk + ks * 32 + g * 8;
  v8h lo = *(const v8h*)(p);
  v8h hi = *(const v8h*)(p + 16);
  return __builtin_shufflevector(lo, hi, 0,1,2,3,4,5,6,7,8,9,10,11,12,13,14,15);
}

// B-fragment from pre-swizzled weights: contiguous 16 halves per lane.
DEVINL v16h load_bfrag(const _Float16* frag, int NT, int ks, int nt, int lane) {
  const _Float16* p = frag + (((size_t)(ks * NT + nt)) * 32 + lane) * 16;
  return *(const v16h*)p;
}

// ---------------------------------------------------------------------------
// Weight prep: frag[ks][nt][lane][h] = w[n][k] * s[n] as f16 (zero pad k>=Cin)
// ---------------------------------------------------------------------------
__global__ void k_prep_w(const float* __restrict__ w, const float* __restrict__ s,
                         _Float16* __restrict__ frag, int Cin, int Cout, int Kpad) {
  int NT = Cout >> 4;
  int total = (Kpad >> 5) * NT * 512;
  int id = blockIdx.x * blockDim.x + threadIdx.x;
  if (id >= total) return;
  int h    = id & 15;
  int lane = (id >> 4) & 31;
  int tile = id >> 9;
  int nt = tile % NT;
  int ks = tile / NT;
  int g = lane >> 4;
  int n = nt * 16 + (lane & 15);
  int k = ks * 32 + ((h & 8) << 1) + g * 8 + (h & 7);
  float v = (k < Cin) ? w[(size_t)n * Cin + k] * s[n] : 0.0f;
  frag[id] = (_Float16)v;
}

// ---------------------------------------------------------------------------
// (B,3,N) -> (B,N,3)
// ---------------------------------------------------------------------------
__global__ void k_transpose(const float* __restrict__ pc, float* __restrict__ xyz,
                            int N, int total) {
  int id = blockIdx.x * blockDim.x + threadIdx.x;
  if (id >= total) return;
  int b = id / N, n = id % N;
  const float* src = pc + (size_t)b * 3 * N + n;
  float* dst = xyz + (size_t)id * 3;
  dst[0] = src[0];
  dst[1] = src[(size_t)N];
  dst[2] = src[(size_t)2 * N];
}

// ---------------------------------------------------------------------------
// Farthest point sampling: one block (256 thr) per batch. Matches jnp.argmax
// first-index tie-breaking.
// ---------------------------------------------------------------------------
__global__ void k_fps(const float* __restrict__ xyz, int N, int npoint,
                      int* __restrict__ outIdx, float* __restrict__ newXyz) {
  __shared__ float dist[4096];
  __shared__ float rv[256];
  __shared__ int   ri[256];
  __shared__ int   sidx[512];
  __shared__ float cur[3];
  int b = blockIdx.x;
  int t = threadIdx.x;
  const float* X = xyz + (size_t)b * N * 3;
  for (int i = t; i < N; i += 256) dist[i] = 1e10f;
  if (t == 0) {
    sidx[0] = 0;
    outIdx[(size_t)b * npoint] = 0;
    cur[0] = X[0]; cur[1] = X[1]; cur[2] = X[2];
  }
  __syncthreads();
  for (int it = 1; it < npoint; ++it) {
    float cx = cur[0], cy = cur[1], cz = cur[2];
    float bestv = -1.0f; int besti = 0;
    for (int i = t; i < N; i += 256) {
      float dx = X[(size_t)i * 3 + 0] - cx;
      float dy = X[(size_t)i * 3 + 1] - cy;
      float dz = X[(size_t)i * 3 + 2] - cz;
      float d = dx * dx + dy * dy + dz * dz;
      float dd = dist[i];
      if (d < dd) { dd = d; dist[i] = d; }
      if (dd > bestv) { bestv = dd; besti = i; }   // strict > keeps lowest index
    }
    rv[t] = bestv; ri[t] = besti;
    __syncthreads();
    for (int off = 128; off > 0; off >>= 1) {
      if (t < off) {
        float v2 = rv[t + off]; int i2 = ri[t + off];
        if (v2 > rv[t] || (v2 == rv[t] && i2 < ri[t])) { rv[t] = v2; ri[t] = i2; }
      }
      __syncthreads();
    }
    if (t == 0) {
      int bi = ri[0];
      sidx[it] = bi;
      outIdx[(size_t)b * npoint + it] = bi;
      cur[0] = X[(size_t)bi * 3 + 0];
      cur[1] = X[(size_t)bi * 3 + 1];
      cur[2] = X[(size_t)bi * 3 + 2];
    }
    __syncthreads();
  }
  for (int i = t; i < npoint; i += 256) {
    int pi = sidx[i];
    float* o = newXyz + ((size_t)b * npoint + i) * 3;
    o[0] = X[(size_t)pi * 3 + 0];
    o[1] = X[(size_t)pi * 3 + 1];
    o[2] = X[(size_t)pi * 3 + 2];
  }
}

// ---------------------------------------------------------------------------
// Ball query: first `nsample` in-radius indices ascending, pad with first.
// One thread per center.
// ---------------------------------------------------------------------------
__global__ void k_ballquery(const float* __restrict__ xyz, const float* __restrict__ newXyz,
                            int N, int S, float r2, int nsample,
                            int* __restrict__ gidx, int total) {
  int id = blockIdx.x * blockDim.x + threadIdx.x;
  if (id >= total) return;
  int b = id / S;
  const float* X = xyz + (size_t)b * N * 3;
  float cx = newXyz[(size_t)id * 3 + 0];
  float cy = newXyz[(size_t)id * 3 + 1];
  float cz = newXyz[(size_t)id * 3 + 2];
  int* out = gidx + (size_t)id * nsample;
  int cnt = 0;
  for (int i = 0; i < N && cnt < nsample; ++i) {
    float dx = X[(size_t)i * 3 + 0] - cx;
    float dy = X[(size_t)i * 3 + 1] - cy;
    float dz = X[(size_t)i * 3 + 2] - cz;
    if (dx * dx + dy * dy + dz * dz <= r2) out[cnt++] = i;
  }
  int f = out[0];  // center is always in-radius -> cnt >= 1
  for (; cnt < nsample; ++cnt) out[cnt] = f;
}

// ---------------------------------------------------------------------------
// Fused SA stage: group (gather - center, concat feats) -> 3x (WMMA GEMM +
// scale-folded bias + relu) -> maxpool over nsample=64. One block (4 waves)
// per center; wave w owns M-tile w.
// ---------------------------------------------------------------------------
__global__ void k_sa_mlp(const float* __restrict__ xyz, const float* __restrict__ newXyz,
                         const int* __restrict__ gidx,
                         const float* __restrict__ featsIn, int Cf, int N, int S,
                         const _Float16* __restrict__ fw0, const float* __restrict__ bi0, int K0, int C1,
                         const _Float16* __restrict__ fw1, const float* __restrict__ bi1, int C2,
                         const _Float16* __restrict__ fw2, const float* __restrict__ bi2, int C3,
                         float* __restrict__ featsOut) {
  __shared__ _Float16 actAbuf[64 * 160];
  __shared__ _Float16 actBbuf[64 * 256];
  int blk = blockIdx.x;
  int b = blk / S;
  int t = threadIdx.x;
  int wave = t >> 5;
  int lane = t & 31;
  const float* X = xyz + (size_t)b * N * 3;
  float cx = newXyz[(size_t)blk * 3 + 0];
  float cy = newXyz[(size_t)blk * 3 + 1];
  float cz = newXyz[(size_t)blk * 3 + 2];
  const int* gi = gidx + (size_t)blk * 64;

  // Build act0 (64 x K0 f16): [xyz - center | feats | zero pad]
  int tot = 64 * K0;
  for (int e = t; e < tot; e += 128) {
    int r = e / K0, k = e % K0;
    int pi = gi[r];
    float v;
    if (k < 3) {
      v = X[(size_t)pi * 3 + k] - (k == 0 ? cx : (k == 1 ? cy : cz));
    } else if (k < 3 + Cf) {
      v = featsIn[((size_t)b * N + pi) * Cf + (k - 3)];
    } else {
      v = 0.0f;
    }
    actAbuf[e] = (_Float16)v;
  }
  __syncthreads();

  _Float16* src = actAbuf;
  _Float16* dst = actBbuf;
  int K = K0;
  int mtile = wave;
  for (int l = 0; l < 3; ++l) {
    const _Float16* fw = (l == 0) ? fw0 : (l == 1) ? fw1 : fw2;
    const float* bias  = (l == 0) ? bi0 : (l == 1) ? bi1 : bi2;
    int Cout           = (l == 0) ? C1  : (l == 1) ? C2  : C3;
    int NT = Cout >> 4;
    int KS = K >> 5;
    for (int nt = 0; nt < NT; ++nt) {
      v8f acc = {0.f, 0.f, 0.f, 0.f, 0.f, 0.f, 0.f, 0.f};
      for (int ks = 0; ks < KS; ++ks) {
        v16h a = load_afrag(src, K, mtile, ks, lane);
        v16h w = load_bfrag(fw, NT, ks, nt, lane);
        acc = wmma_f32_f16(a, w, acc);
      }
      int n = nt * 16 + (lane & 15);
      int g = lane >> 4;
      float bv = bias[n];
#pragma unroll
      for (int j = 0; j < 8; ++j) {
        int r = mtile * 16 + j + 8 * g;
        float v = acc[j] + bv;
        v = v > 0.0f ? v : 0.0f;
        dst[(size_t)r * Cout + n] = (_Float16)v;
      }
    }
    __syncthreads();
    _Float16* tmp = src; src = dst; dst = tmp;
    K = Cout;
  }

  // maxpool over 64 rows (src now holds 64 x C3, relu'd so >= 0)
  for (int c = t; c < C3; c += 128) {
    float m = 0.0f;
    for (int r = 0; r < 64; ++r) {
      float v = (float)src[(size_t)r * C3 + c];
      m = v > m ? v : m;
    }
    featsOut[(size_t)blk * C3 + c] = m;
  }
}

// ---------------------------------------------------------------------------
// SA3 input: act0[b][r][0:3]=xyz2, [3:259]=feats2, pad to 288, f16
// ---------------------------------------------------------------------------
__global__ void k_build_sa3(const float* __restrict__ xyz2, const float* __restrict__ feats2,
                            _Float16* __restrict__ a0, int total) {
  int id = blockIdx.x * blockDim.x + threadIdx.x;
  if (id >= total) return;
  int k = id % 288;
  int row = id / 288;  // row = b*128 + r
  float v;
  if (k < 3)        v = xyz2[(size_t)row * 3 + k];
  else if (k < 259) v = feats2[(size_t)row * 256 + (k - 3)];
  else              v = 0.0f;
  a0[id] = (_Float16)v;
}

// ---------------------------------------------------------------------------
// Generic WMMA GEMM + bias + relu, one wave per (batch, mtile, ntile).
// ---------------------------------------------------------------------------
__global__ void k_gemm_bias_relu(const _Float16* __restrict__ act, const _Float16* __restrict__ fragW,
                                 const float* __restrict__ bias, _Float16* __restrict__ out,
                                 int M, int K, int Cout) {
  int NT = Cout >> 4, MT = M >> 4;
  int id = blockIdx.x;
  int nt = id % NT; id /= NT;
  int mt = id % MT;
  int b  = id / MT;
  int lane = threadIdx.x & 31;
  const _Float16* A = act + (size_t)b * M * K;
  _Float16* O = out + (size_t)b * M * Cout;
  int KS = K >> 5;
  v8f acc = {0.f, 0.f, 0.f, 0.f, 0.f, 0.f, 0.f, 0.f};
  for (int ks = 0; ks < KS; ++ks) {
    v16h a = load_afrag(A, K, mt, ks, lane);
    v16h w = load_bfrag(fragW, NT, ks, nt, lane);
    acc = wmma_f32_f16(a, w, acc);
  }
  int n = nt * 16 + (lane & 15);
  int g = lane >> 4;
  float bv = bias[n];
#pragma unroll
  for (int j = 0; j < 8; ++j) {
    int r = mt * 16 + j + 8 * g;
    float v = acc[j] + bv;
    v = v > 0.0f ? v : 0.0f;
    O[(size_t)r * Cout + n] = (_Float16)v;
  }
}

// ---------------------------------------------------------------------------
// Column-wise max over M rows: out[b][c] = max_r act[b][r][c]
// ---------------------------------------------------------------------------
__global__ void k_maxpool_rows(const _Float16* __restrict__ act, float* __restrict__ out,
                               int M, int C, int total) {
  int id = blockIdx.x * blockDim.x + threadIdx.x;
  if (id >= total) return;
  int b = id / C, c = id % C;
  const _Float16* A = act + (size_t)b * M * C + c;
  float m = 0.0f;
  for (int r = 0; r < M; ++r) {
    float v = (float)A[(size_t)r * C];
    m = v > m ? v : m;
  }
  out[id] = m;
}

// ---------------------------------------------------------------------------
// Final FC head (full fp32): h = relu(f@w1.T * s1 + b1); out = h@w2.T + b2
// One block per batch.
// ---------------------------------------------------------------------------
__global__ void k_fc(const float* __restrict__ f, const float* __restrict__ w1,
                     const float* __restrict__ s1, const float* __restrict__ b1,
                     const float* __restrict__ w2, const float* __restrict__ b2,
                     float* __restrict__ out) {
  __shared__ float sf[1024];
  __shared__ float sh[512];
  int b = blockIdx.x;
  int t = threadIdx.x;
  for (int i = t; i < 1024; i += 256) sf[i] = f[(size_t)b * 1024 + i];
  __syncthreads();
  for (int c = t; c < 512; c += 256) {
    const float* w = w1 + (size_t)c * 1024;
    float acc = 0.0f;
    for (int i = 0; i < 1024; ++i) acc += sf[i] * w[i];
    acc = acc * s1[c] + b1[c];
    sh[c] = acc > 0.0f ? acc : 0.0f;
  }
  __syncthreads();
  for (int c = t; c < 256; c += 256) {
    const float* w = w2 + (size_t)c * 512;
    float acc = 0.0f;
    for (int i = 0; i < 512; ++i) acc += sh[i] * w[i];
    out[(size_t)b * 256 + c] = acc + b2[c];
  }
}

// ---------------------------------------------------------------------------
// Host launcher
// ---------------------------------------------------------------------------
extern "C" void kernel_launch(void* const* d_in, const int* in_sizes, int n_in,
                              void* d_out, int out_size, void* d_ws, size_t ws_size,
                              hipStream_t stream) {
  (void)in_sizes; (void)n_in; (void)out_size; (void)ws_size;
  const int B = 16, N = 4096;

  // Inputs in setup_inputs() dict insertion order:
  // 0: pointcloud; then sa1/sa2/sa3 layers as (w,s,b) triples; then fc w1,s1,b1,w2,b2
  const float* pc = (const float*)d_in[0];
  const float *W[9], *SC[9], *BI[9];
  for (int l = 0; l < 9; ++l) {
    W[l]  = (const float*)d_in[1 + 3 * l];
    SC[l] = (const float*)d_in[2 + 3 * l];
    BI[l] = (const float*)d_in[3 + 3 * l];
  }
  const float* fw1 = (const float*)d_in[28];
  const float* fs1 = (const float*)d_in[29];
  const float* fb1 = (const float*)d_in[30];
  const float* fw2 = (const float*)d_in[31];
  const float* fb2 = (const float*)d_in[32];
  float* outp = (float*)d_out;

  // Workspace bump allocator (256B aligned)
  char* wp = (char*)d_ws;
  auto alloc = [&](size_t bytes) -> char* {
    char* r = wp;
    wp += (bytes + 255) & ~(size_t)255;
    return r;
  };

  float*    xyz    = (float*)alloc((size_t)B * N * 3 * 4);
  int*      idx1   = (int*)alloc((size_t)B * 512 * 4);
  float*    nxyz1  = (float*)alloc((size_t)B * 512 * 3 * 4);
  int*      gidx1  = (int*)alloc((size_t)B * 512 * 64 * 4);
  float*    feats1 = (float*)alloc((size_t)B * 512 * 128 * 4);
  int*      idx2   = (int*)alloc((size_t)B * 128 * 4);
  float*    nxyz2  = (float*)alloc((size_t)B * 128 * 3 * 4);
  int*      gidx2  = (int*)alloc((size_t)B * 128 * 64 * 4);
  float*    feats2 = (float*)alloc((size_t)B * 128 * 256 * 4);
  _Float16* a3_0   = (_Float16*)alloc((size_t)B * 128 * 288 * 2);
  _Float16* a3_1   = (_Float16*)alloc((size_t)B * 128 * 256 * 2);
  _Float16* a3_2   = (_Float16*)alloc((size_t)B * 128 * 512 * 2);
  _Float16* a3_3   = (_Float16*)alloc((size_t)B * 128 * 1024 * 2);
  float*    fvec   = (float*)alloc((size_t)B * 1024 * 4);

  static const int Cin[9]  = {3, 64, 64, 131, 128, 128, 259, 256, 512};
  static const int Cout[9] = {64, 64, 128, 128, 128, 256, 256, 512, 1024};
  static const int Kpad[9] = {32, 64, 64, 160, 128, 128, 288, 256, 512};
  _Float16* frag[9];
  for (int l = 0; l < 9; ++l)
    frag[l] = (_Float16*)alloc((size_t)Kpad[l] * Cout[l] * 2);

  // ---- preprocessing ----
  {
    int total = B * N;
    k_transpose<<<(total + 255) / 256, 256, 0, stream>>>(pc, xyz, N, total);
  }
  for (int l = 0; l < 9; ++l) {
    int total = Kpad[l] * Cout[l];
    k_prep_w<<<(total + 255) / 256, 256, 0, stream>>>(W[l], SC[l], frag[l],
                                                      Cin[l], Cout[l], Kpad[l]);
  }

  // ---- SA1 ----
  k_fps<<<B, 256, 0, stream>>>(xyz, N, 512, idx1, nxyz1);
  {
    int total = B * 512;
    k_ballquery<<<(total + 255) / 256, 256, 0, stream>>>(xyz, nxyz1, N, 512,
                                                         0.2f * 0.2f, 64, gidx1, total);
  }
  k_sa_mlp<<<B * 512, 128, 0, stream>>>(xyz, nxyz1, gidx1,
                                        (const float*)nullptr, 0, N, 512,
                                        frag[0], BI[0], Kpad[0], Cout[0],
                                        frag[1], BI[1], Cout[1],
                                        frag[2], BI[2], Cout[2],
                                        feats1);

  // ---- SA2 ----
  k_fps<<<B, 256, 0, stream>>>(nxyz1, 512, 128, idx2, nxyz2);
  {
    int total = B * 128;
    k_ballquery<<<(total + 255) / 256, 256, 0, stream>>>(nxyz1, nxyz2, 512, 128,
                                                         0.4f * 0.4f, 64, gidx2, total);
  }
  k_sa_mlp<<<B * 128, 128, 0, stream>>>(nxyz1, nxyz2, gidx2,
                                        feats1, 128, 512, 128,
                                        frag[3], BI[3], Kpad[3], Cout[3],
                                        frag[4], BI[4], Cout[4],
                                        frag[5], BI[5], Cout[5],
                                        feats2);

  // ---- SA3 (group-all): 3 WMMA GEMMs + maxpool ----
  {
    int total = B * 128 * 288;
    k_build_sa3<<<(total + 255) / 256, 256, 0, stream>>>(nxyz2, feats2, a3_0, total);
  }
  k_gemm_bias_relu<<<B * (128 / 16) * (256 / 16), 32, 0, stream>>>(
      a3_0, frag[6], BI[6], a3_1, 128, 288, 256);
  k_gemm_bias_relu<<<B * (128 / 16) * (512 / 16), 32, 0, stream>>>(
      a3_1, frag[7], BI[7], a3_2, 128, 256, 512);
  k_gemm_bias_relu<<<B * (128 / 16) * (1024 / 16), 32, 0, stream>>>(
      a3_2, frag[8], BI[8], a3_3, 128, 512, 1024);
  {
    int total = B * 1024;
    k_maxpool_rows<<<(total + 255) / 256, 256, 0, stream>>>(a3_3, fvec, 128, 1024, total);
  }

  // ---- FC head ----
  k_fc<<<B, 256, 0, stream>>>(fvec, fw1, fs1, fb1, fw2, fb2, outp);
}
